// Model_44126493999720
// MI455X (gfx1250) — matrix operand order repaired and checked by reading, not compiled, not run on hardware
//
#include <hip/hip_runtime.h>

// ---------------- types for WMMA ----------------
typedef __attribute__((ext_vector_type(16))) __bf16 v16bf;
typedef __attribute__((ext_vector_type(8)))  float  v8f;

union FragBF { unsigned int u[8]; v16bf v; };

// ---------------- bf16 helpers ----------------
__device__ __forceinline__ unsigned short f2bf(float f) {
    unsigned int u = __float_as_uint(f);
    u = u + 0x7FFFu + ((u >> 16) & 1u);   // round-to-nearest-even
    return (unsigned short)(u >> 16);
}
__device__ __forceinline__ float bf2f(unsigned short h) {
    return __uint_as_float(((unsigned int)h) << 16);
}

#define EPS 1e-5f

// =====================================================================
// GEMM: C[M,128] (f32) = A[M,K] (bf16, ld=K) x W[K,128] (bf16)
// Macro-tile 128x128, 8 waves/block, each wave computes 16 rows x 128 cols.
// K must be a multiple of 32. Software-pipelined double-buffered LDS.
// =====================================================================
__global__ __launch_bounds__(256) void gemm_bf16(
    const unsigned short* __restrict__ A,
    const unsigned short* __restrict__ W,
    float* __restrict__ Y, int M, int K)
{
    __shared__ unsigned short As[2][128 * 32];   // row-major [r][k]
    __shared__ unsigned short Bs[2][128 * 32];   // K-major transposed [col][k]

    const int tid  = threadIdx.x;
    const int lane = tid & 31;
    const int wave = tid >> 5;
    const int mBase = blockIdx.x * 128;

    const int r0 = lane & 15;
    const int hi = lane >> 4;

    v8f acc[8];
    #pragma unroll
    for (int n = 0; n < 8; n++)
        #pragma unroll
        for (int j = 0; j < 8; j++) acc[n][j] = 0.0f;

    // staging registers for the next tile (global -> regs -> LDS)
    uint4        aReg[2];
    unsigned int bReg[8];

    auto loadGlobal = [&](int k0) {
        #pragma unroll
        for (int it = 0; it < 2; ++it) {
            int idx = tid + it * 256;          // 0..511
            int r   = idx >> 2;                // 0..127
            int c8  = (idx & 3) << 3;          // 0,8,16,24
            uint4 v = {0u, 0u, 0u, 0u};
            int gr = mBase + r;
            if (gr < M)
                v = *(const uint4*)(A + (size_t)gr * K + k0 + c8);
            aReg[it] = v;
        }
        #pragma unroll
        for (int it = 0; it < 8; ++it) {
            int idx = tid + it * 256;          // 0..2047
            int kr  = idx >> 6;                // 0..31
            int cp  = (idx & 63) << 1;         // even column
            bReg[it] = *(const unsigned int*)(W + (size_t)(k0 + kr) * 128 + cp);
        }
    };
    auto storeLDS = [&](int buf) {
        #pragma unroll
        for (int it = 0; it < 2; ++it) {
            int idx = tid + it * 256;
            int r   = idx >> 2;
            int c8  = (idx & 3) << 3;
            *(uint4*)(&As[buf][r * 32 + c8]) = aReg[it];
        }
        #pragma unroll
        for (int it = 0; it < 8; ++it) {
            int idx = tid + it * 256;
            int kr  = idx >> 6;
            int cp  = (idx & 63) << 1;
            Bs[buf][cp * 32 + kr]       = (unsigned short)(bReg[it] & 0xFFFFu);
            Bs[buf][(cp + 1) * 32 + kr] = (unsigned short)(bReg[it] >> 16);
        }
    };

    // prologue: stage K-tile 0
    loadGlobal(0);
    storeLDS(0);
    int buf = 0;

    for (int k0 = 0; k0 < K; k0 += 32) {
        __syncthreads();
        const bool more = (k0 + 32) < K;
        // issue global loads for next tile early; latency hidden by WMMAs
        if (more) loadGlobal(k0 + 32);
        // prefetch A tile two steps ahead into GL2 (gfx1250 global_prefetch_b8)
        if (k0 + 64 < K) {
            int r  = tid >> 2;
            int c8 = (tid & 3) << 3;
            int gr = mBase + r;
            if (gr < M)
                __builtin_prefetch(A + (size_t)gr * K + k0 + 64 + c8, 0, 1);
        }

        // ---- A fragment (16x32 bf16 per wave); ISA A layout: k = 2p + (p>=4?8:0) + 8*hi
        FragBF af;
        const int ar = wave * 16 + r0;
        #pragma unroll
        for (int p = 0; p < 8; p++) {
            int kb = 2 * p + ((p >= 4) ? 8 : 0) + hi * 8;
            af.u[p] = *(const unsigned int*)(&As[buf][ar * 32 + kb]);
        }
        // ---- preload all 8 B fragments; ISA B layout: k = 2p + (p>=4?8:0) + 8*hi
        FragBF bfr[8];
        #pragma unroll
        for (int n = 0; n < 8; n++) {
            const int bc = n * 16 + r0;
            #pragma unroll
            for (int p = 0; p < 8; p++) {
                int kb = 2 * p + ((p >= 4) ? 8 : 0) + hi * 8;
                bfr[n].u[p] = *(const unsigned int*)(&Bs[buf][bc * 32 + kb]);
            }
        }
        // ---- 8 back-to-back WMMAs ----
        #pragma unroll
        for (int n = 0; n < 8; n++) {
            acc[n] = __builtin_amdgcn_wmma_f32_16x16x32_bf16(
                false, af.v, false, bfr[n].v, (short)0, acc[n], false, false);
        }

        // write next tile into the other LDS buffer after compute
        if (more) { storeLDS(buf ^ 1); buf ^= 1; }
    }

    // ---- store C (f32 16x16 layout: lane<16 -> M=i, lane>=16 -> M=8+i) ----
    const int mrow = mBase + wave * 16;
    #pragma unroll
    for (int n = 0; n < 8; n++) {
        int c = n * 16 + r0;
        #pragma unroll
        for (int i = 0; i < 8; i++) {
            int r = mrow + (hi ? (8 + i) : i);
            if (r < M) Y[(size_t)r * 128 + c] = acc[n][i];
        }
    }
}

// =====================================================================
// Column stats: stats[c] += sum, stats[C+c] += sumsq  (atomic partials)
// =====================================================================
__global__ void col_stats(const float* __restrict__ Y, int M, int C,
                          float* __restrict__ stats)
{
    int c = threadIdx.x;
    if (c >= C) return;
    float s = 0.f, s2 = 0.f;
    for (int r = blockIdx.x; r < M; r += gridDim.x) {
        float v = Y[(size_t)r * C + c];
        s += v; s2 += v * v;
    }
    atomicAdd(&stats[c], s);
    atomicAdd(&stats[C + c], s2);
}

// =====================================================================
// BN (training, gamma=1,beta=0) + optional PReLU; dual-format output
// =====================================================================
__global__ void bn_prelu(const float* __restrict__ Y, int M, int C,
                         const float* __restrict__ stats,
                         const float* __restrict__ alpha,
                         float* __restrict__ of32,
                         unsigned short* __restrict__ obf16, int ldo)
{
    int t = blockIdx.x * 256 + threadIdx.x;
    if (t >= M * C) return;
    int m = t / C, c = t - m * C;
    float invM = 1.0f / (float)M;
    float mu  = stats[c] * invM;
    float var = stats[C + c] * invM - mu * mu;
    float y = (Y[t] - mu) * rsqrtf(var + EPS);
    if (alpha) { float a = alpha[0]; y = (y > 0.f) ? y : a * y; }
    if (of32)  of32[t] = y;
    if (obf16) obf16[(size_t)m * ldo + c] = f2bf(y);
}

// =====================================================================
// scatter add: acc[idx[m]][0:128] += src[m][0:128]; cnt[idx[m]] += 1
// =====================================================================
__global__ void scatter_add(const float* __restrict__ src,
                            const int* __restrict__ idx,
                            float* __restrict__ acc,
                            float* __restrict__ cnt, int M)
{
    int t = blockIdx.x * 256 + threadIdx.x;
    int m = t >> 5, q = t & 31;
    if (m >= M) return;
    int d = idx[m];
    #pragma unroll
    for (int j = 0; j < 4; j++)
        atomicAdd(&acc[(size_t)d * 128 + q * 4 + j], src[(size_t)m * 128 + q * 4 + j]);
    if (q == 0) atomicAdd(&cnt[d], 1.0f);
}

__global__ void divide_k(float* __restrict__ acc, const float* __restrict__ cnt, int total)
{
    int t = blockIdx.x * 256 + threadIdx.x;
    if (t < total) acc[t] /= fmaxf(cnt[t >> 7], 1.0f);
}

// =====================================================================
// Input builders (write bf16 GEMM A matrices, zero-padded to Kp)
// =====================================================================
__global__ void embed_x_k(const int* __restrict__ x_cat, const float* __restrict__ x_feat,
                          const float* __restrict__ emb, float* __restrict__ outp, int Nn)
{
    int t = blockIdx.x * 256 + threadIdx.x;
    if (t >= Nn * 15) return;
    int n = t / 15, j = t - n * 15;
    outp[t] = (j < 8) ? emb[x_cat[n] * 8 + j] : x_feat[n * 7 + (j - 8)];
}

__global__ void embed_e_k(const int* __restrict__ e_cat, const float* __restrict__ e_feat,
                          const float* __restrict__ emb, float* __restrict__ outp, int Ee)
{
    int t = blockIdx.x * 256 + threadIdx.x;
    if (t >= Ee * 12) return;
    int e = t / 12, j = t - e * 12;
    outp[t] = (j < 8) ? emb[e_cat[e] * 8 + j] : e_feat[e * 4 + (j - 8)];
}

__global__ void build_edge_in(unsigned short* __restrict__ A,
    const unsigned short* __restrict__ x, int ldx, int xc,
    const unsigned short* __restrict__ e, int lde, int ec,
    const unsigned short* __restrict__ g, int ldg, int gc,
    const int* __restrict__ row, const int* __restrict__ colI,
    const int* __restrict__ batch, int eoff, int goff, int Kp, int Ee)
{
    int t = blockIdx.x * 256 + threadIdx.x;
    int ed = t / Kp, k = t - ed * Kp;
    if (ed >= Ee) return;
    unsigned short v = 0;
    if (k < xc) {
        int r = row[ed], c = colI[ed];
        v = f2bf(0.5f * (bf2f(x[(size_t)r * ldx + k]) + bf2f(x[(size_t)c * ldx + k])));
    } else if (k >= eoff && k < eoff + ec) {
        v = e[(size_t)ed * lde + (k - eoff)];
    } else if (k >= goff && k < goff + gc) {
        v = g[(size_t)batch[row[ed]] * ldg + (k - goff)];
    }
    A[(size_t)ed * Kp + k] = v;
}

__global__ void build_node1_in(unsigned short* __restrict__ A,
    const unsigned short* __restrict__ x, int ldx, int xc,
    const int* __restrict__ colI,
    const unsigned short* __restrict__ e, int eoff, int Kp, int Ee)
{
    int t = blockIdx.x * 256 + threadIdx.x;
    int ed = t / Kp, k = t - ed * Kp;
    if (ed >= Ee) return;
    unsigned short v = 0;
    if (k < xc)                              v = x[(size_t)colI[ed] * ldx + k];
    else if (k >= eoff && k < eoff + 128)    v = e[(size_t)ed * 128 + (k - eoff)];
    A[(size_t)ed * Kp + k] = v;
}

__global__ void build_node2_in(unsigned short* __restrict__ A,
    const float* __restrict__ h,
    const unsigned short* __restrict__ g, int ldg, int gc,
    const int* __restrict__ batch, int Kp, int Nn)
{
    int t = blockIdx.x * 256 + threadIdx.x;
    int n = t / Kp, k = t - n * Kp;
    if (n >= Nn) return;
    unsigned short v = 0;
    if (k < 128)                        v = f2bf(h[(size_t)n * 128 + k]);
    else if (k >= 128 && k < 128 + gc)  v = g[(size_t)batch[n] * ldg + (k - 128)];
    A[(size_t)n * Kp + k] = v;
}

__global__ void build_glob_in(unsigned short* __restrict__ A,
    const unsigned short* __restrict__ g, int ldg, int gc,
    const float* __restrict__ xm, int xoff, int Kp, int Bb)
{
    int t = blockIdx.x * 256 + threadIdx.x;
    int b = t / Kp, k = t - b * Kp;
    if (b >= Bb) return;
    unsigned short v = 0;
    if (k < gc)                           v = g[(size_t)b * ldg + k];
    else if (k >= xoff && k < xoff + 128) v = f2bf(xm[(size_t)b * 128 + (k - xoff)]);
    A[(size_t)b * Kp + k] = v;
}

// pad/convert weights: src rows are concatenated segments l0,l1,l2 placed at
// destination row offsets d0,d1,d2 in a [Kp,128] bf16 matrix, rest zero.
__global__ void pad_w(const float* __restrict__ W, unsigned short* __restrict__ Wd,
                      int Kp, int l0, int d0, int l1, int d1, int l2, int d2)
{
    int t = blockIdx.x * 256 + threadIdx.x;
    if (t >= Kp * 128) return;
    int r = t >> 7, c = t & 127;
    float v = 0.f;
    if (r >= d0 && r < d0 + l0)              v = W[(r - d0) * 128 + c];
    else if (l1 && r >= d1 && r < d1 + l1)   v = W[(l0 + r - d1) * 128 + c];
    else if (l2 && r >= d2 && r < d2 + l2)   v = W[(l0 + l1 + r - d2) * 128 + c];
    Wd[t] = f2bf(v);
}

// =====================================================================
// output head: o1 = e@o1_w + b1 ; o2 = o1@o2_w + b2 ; out = [o2, o1]
// =====================================================================
__global__ void final_out(const unsigned short* __restrict__ e,
                          const float* __restrict__ o1w, const float* __restrict__ o1b,
                          const float* __restrict__ o2w, const float* __restrict__ o2b,
                          float* __restrict__ outp, int Ee)
{
    int ed = blockIdx.x * 256 + threadIdx.x;
    if (ed >= Ee) return;
    float o1[4] = {o1b[0], o1b[1], o1b[2], o1b[3]};
    const unsigned short* er = e + (size_t)ed * 128;
    for (int k = 0; k < 128; k++) {
        float xv = bf2f(er[k]);
        #pragma unroll
        for (int j = 0; j < 4; j++) o1[j] += xv * o1w[k * 4 + j];
    }
    float o2 = o2b[0];
    #pragma unroll
    for (int j = 0; j < 4; j++) o2 += o1[j] * o2w[j];
    outp[(size_t)ed * 5] = o2;
    #pragma unroll
    for (int j = 0; j < 4; j++) outp[(size_t)ed * 5 + 1 + j] = o1[j];
}

// =====================================================================
// host launcher
// =====================================================================
extern "C" void kernel_launch(void* const* d_in, const int* in_sizes, int n_in,
                              void* d_out, int out_size, void* d_ws, size_t ws_size,
                              hipStream_t stream)
{
    (void)in_sizes; (void)n_in; (void)out_size; (void)ws_size;
    const int N = 16384, E = 262144, B = 64;

    const int*   x_cat   = (const int*)  d_in[0];
    const float* x_feat  = (const float*)d_in[1];
    const int*   row     = (const int*)  d_in[2];
    const int*   colI    = row + E;
    const int*   e_cat   = (const int*)  d_in[3];
    const float* e_feat  = (const float*)d_in[4];
    const float* u       = (const float*)d_in[5];
    const int*   batch   = (const int*)  d_in[6];
    const float* node_emb= (const float*)d_in[7];
    const float* edge_emb= (const float*)d_in[8];
    const float* edge_w[4]; for (int i = 0; i < 4; i++) edge_w[i] = (const float*)d_in[9 + i];
    const float* nw1[3];    for (int i = 0; i < 3; i++) nw1[i]    = (const float*)d_in[13 + i];
    const float* nw2[3];    for (int i = 0; i < 3; i++) nw2[i]    = (const float*)d_in[16 + i];
    const float* gw[3];     for (int i = 0; i < 3; i++) gw[i]     = (const float*)d_in[19 + i];
    const float* pe[4];     for (int i = 0; i < 4; i++) pe[i]     = (const float*)d_in[22 + i];
    const float* pn1[3];    for (int i = 0; i < 3; i++) pn1[i]    = (const float*)d_in[26 + i];
    const float* pn2[3];    for (int i = 0; i < 3; i++) pn2[i]    = (const float*)d_in[29 + i];
    const float* pg[3];     for (int i = 0; i < 3; i++) pg[i]     = (const float*)d_in[32 + i];
    const float* o1w = (const float*)d_in[35];
    const float* o1b = (const float*)d_in[36];
    const float* o2w = (const float*)d_in[37];
    const float* o2b = (const float*)d_in[38];
    float* outp = (float*)d_out;

    // ---- workspace carve ----
    char* ws = (char*)d_ws;
    size_t off = 0;
    auto alloc = [&](size_t bytes) -> char* {
        char* p = ws + off;
        off = (off + bytes + 255) & ~(size_t)255;
        return p;
    };
    unsigned short* A_buf = (unsigned short*)alloc((size_t)E * 384 * 2);
    float*          Y_buf = (float*)         alloc((size_t)E * 128 * 4);
    unsigned short* e_bf  = (unsigned short*)alloc((size_t)E * 128 * 2);
    unsigned short* x_bf  = (unsigned short*)alloc((size_t)N * 128 * 2);
    unsigned short* g_bf  = (unsigned short*)alloc((size_t)B * 128 * 2);
    float*          nacc  = (float*)         alloc((size_t)N * 128 * 4 + N * 4);
    float*          ncnt  = nacc + (size_t)N * 128;
    float*          gacc  = (float*)         alloc((size_t)B * 128 * 4 + B * 4);
    float*          gcnt  = gacc + (size_t)B * 128;
    float*          stats = (float*)         alloc(1024);
    unsigned short* Wpad  = (unsigned short*)alloc((size_t)512 * 128 * 2);
    float* x_raw = nacc;   // reused before nacc is needed
    float* e_raw = Y_buf;  // reused before Y_buf is needed

    auto bn = [&](const float* Yp, int M, int C, const float* a,
                  float* of32, unsigned short* ob, int ldo) {
        hipMemsetAsync(stats, 0, 2 * C * sizeof(float), stream);
        col_stats<<<dim3(512), dim3(128), 0, stream>>>(Yp, M, C, stats);
        int tot = M * C;
        bn_prelu<<<dim3((tot + 255) / 256), dim3(256), 0, stream>>>(Yp, M, C, stats, a, of32, ob, ldo);
    };
    auto gemm = [&](const unsigned short* Ap, int M, int K) {
        gemm_bf16<<<dim3((M + 127) / 128), dim3(256), 0, stream>>>(Ap, Wpad, Y_buf, M, K);
    };
    auto padw = [&](const float* Wsrc, int Kp, int l0, int d0, int l1, int d1, int l2, int d2) {
        pad_w<<<dim3((Kp * 128 + 255) / 256), dim3(256), 0, stream>>>(Wsrc, Wpad, Kp, l0, d0, l1, d1, l2, d2);
    };

    // ---- initial embeds + BN ----
    embed_x_k<<<dim3((N * 15 + 255) / 256), dim3(256), 0, stream>>>(x_cat, x_feat, node_emb, x_raw, N);
    bn(x_raw, N, 15, nullptr, nullptr, x_bf, 15);
    embed_e_k<<<dim3((E * 12 + 255) / 256), dim3(256), 0, stream>>>(e_cat, e_feat, edge_emb, e_raw, E);
    bn(e_raw, E, 12, nullptr, nullptr, e_bf, 12);
    bn(u, B, 14, nullptr, nullptr, g_bf, 14);

    int xc = 15, xld = 15, ec = 12, eld = 12, gc = 14, gld = 14;

    for (int i = 0; i < 4; i++) {
        // ---------- edge update ----------
        int eoff = (i == 0) ? 16 : 128, goff = (i == 0) ? 32 : 256, Kp = (i == 0) ? 64 : 384;
        build_edge_in<<<dim3((E * Kp + 255) / 256), dim3(256), 0, stream>>>(
            A_buf, x_bf, xld, xc, e_bf, eld, ec, g_bf, gld, gc, row, colI, batch, eoff, goff, Kp, E);
        padw(edge_w[i], Kp, xc, 0, ec, eoff, gc, goff);
        gemm(A_buf, E, Kp);
        bn(Y_buf, E, 128, pe[i], nullptr, e_bf, 128);
        eld = 128; ec = 128;

        if (i < 3) {
            // ---------- node update ----------
            int Kp1 = (i == 0) ? 160 : 256, eo1 = (i == 0) ? 16 : 128;
            build_node1_in<<<dim3((E * Kp1 + 255) / 256), dim3(256), 0, stream>>>(
                A_buf, x_bf, xld, xc, colI, e_bf, eo1, Kp1, E);
            padw(nw1[i], Kp1, xc, 0, 128, eo1, 0, 0);
            gemm(A_buf, E, Kp1);
            bn(Y_buf, E, 128, pn1[i], Y_buf, nullptr, 0);

            hipMemsetAsync(nacc, 0, (size_t)N * 128 * 4 + N * 4, stream);
            scatter_add<<<dim3((E * 32 + 255) / 256), dim3(256), 0, stream>>>(Y_buf, row, nacc, ncnt, E);
            divide_k<<<dim3((N * 128 + 255) / 256), dim3(256), 0, stream>>>(nacc, ncnt, N * 128);

            int Kp2 = (i == 0) ? 160 : 256;
            build_node2_in<<<dim3((N * Kp2 + 255) / 256), dim3(256), 0, stream>>>(
                A_buf, nacc, g_bf, gld, gc, batch, Kp2, N);
            padw(nw2[i], Kp2, 128, 0, gc, 128, 0, 0);
            gemm(A_buf, N, Kp2);
            bn(Y_buf, N, 128, pn2[i], Y_buf, x_bf, 128);
            xld = 128; xc = 128;

            // ---------- global update ----------
            hipMemsetAsync(gacc, 0, (size_t)B * 128 * 4 + B * 4, stream);
            scatter_add<<<dim3((N * 32 + 255) / 256), dim3(256), 0, stream>>>(Y_buf, batch, gacc, gcnt, N);
            divide_k<<<dim3((B * 128 + 255) / 256), dim3(256), 0, stream>>>(gacc, gcnt, B * 128);

            int Kp3 = (i == 0) ? 160 : 256, xo3 = (i == 0) ? 16 : 128;
            build_glob_in<<<dim3((B * Kp3 + 255) / 256), dim3(256), 0, stream>>>(
                A_buf, g_bf, gld, gc, gacc, xo3, Kp3, B);
            padw(gw[i], Kp3, gc, 0, 128, xo3, 0, 0);
            gemm(A_buf, B, Kp3);
            bn(Y_buf, B, 128, pg[i], nullptr, g_bf, 128);
            gld = 128; gc = 128;
        }
    }

    // ---- output head ----
    final_out<<<dim3((E + 255) / 256), dim3(256), 0, stream>>>(e_bf, o1w, o1b, o2w, o2b, outp, E);
}
